// HierarchicalSoftmax_47236050321601
// MI455X (gfx1250) — matrix operand order repaired and checked by reading, not compiled, not run on hardware
//
#include <hip/hip_runtime.h>
#include <hip/hip_bf16.h>

#define B_    128
#define H_    1024
#define V_    32000
#define NW    31999            // weight rows (internal nodes)
#define TN    128              // N tile per block
#define KS    64               // K per TDM stage (= 2 WMMA K-steps)
#define NSTAGE (H_ / KS)       // 16
#define MAXD  24

// LDS layout (dynamic):
//   A tile : 128 rows x (1024+8) f16, stride 2064 B  -> 264192 B
//   W bufs : 2 x (128 rows x 144 B)                  -> 2 x 18432 B
#define A_STRIDE_B   2064
#define A_BYTES      (128 * A_STRIDE_B)
#define W_STRIDE_B   144
#define W_TILE_B     (128 * W_STRIDE_B)
#define W0_OFF       A_BYTES
#define SMEM_BYTES   (A_BYTES + 2 * W_TILE_B)   // 301056 B < 320 KB

typedef float    v8f  __attribute__((ext_vector_type(8)));
typedef _Float16 v16h __attribute__((ext_vector_type(16)));
typedef _Float16 v8h  __attribute__((ext_vector_type(8)));

// ---------------- CDNA5 async global->LDS (ASYNCcnt) ----------------
__device__ __forceinline__ void async_b128(const void* g, const void* lds_ptr) {
    unsigned lds = (unsigned)(unsigned long long)lds_ptr; // low 32 bits = LDS offset
    asm volatile("global_load_async_to_lds_b128 %0, %1, off"
                 :: "v"(lds), "v"(g) : "memory");
}
__device__ __forceinline__ void wait_async0() {
#if __has_builtin(__builtin_amdgcn_s_wait_asynccnt)
    __builtin_amdgcn_s_wait_asynccnt(0);
#else
    asm volatile("s_wait_asynccnt 0" ::: "memory");
#endif
}
__device__ __forceinline__ void wait_async4() {
    asm volatile("s_wait_asynccnt 4" ::: "memory");
}

// ---------------- CDNA5 Tensor Data Mover ----------------
#if defined(__has_builtin)
#  if __has_builtin(__builtin_amdgcn_tensor_load_to_lds)
#    define HAVE_TDM 1
#  endif
#endif
#ifndef HAVE_TDM
#  define HAVE_TDM 0
#endif
#if HAVE_TDM && defined(__has_include)
#  if __has_include(<hip/amd_detail/amd_gfx1250_TDM.h>)
#    define TDM_6ARG 1
#  endif
#endif
#ifndef TDM_6ARG
#  define TDM_6ARG 0
#endif

#if HAVE_TDM
typedef unsigned int v4u __attribute__((ext_vector_type(4)));
typedef int          v4i __attribute__((ext_vector_type(4)));
typedef int          v8i __attribute__((ext_vector_type(8)));

// 2D f16 tile load: tile_k=64 (row = 128 B), tile_rows=128, row stride 1024
// elems, with LDS padding of 16 B after every 128 B (pad_interval=4 -> 16
// qwords, pad_amount=3 -> 4 dwords) => 144 B LDS row stride, bank-spread.
// rem_k / rem_rows: remaining tensor extent from tile origin (OOB reads -> 0).
__device__ __forceinline__ void tdm_load_w_tile(const void* gtile, unsigned ldsoff,
                                                unsigned rem_k, unsigned rem_rows) {
    unsigned long long ga = (unsigned long long)gtile;
    v4u g0;
    g0[0] = 1u;                                             // count=1 (user D#)
    g0[1] = ldsoff;                                         // lds_addr (bytes)
    g0[2] = (unsigned)ga;                                   // global_addr[31:0]
    g0[3] = (unsigned)((ga >> 32) & 0x01FFFFFFu) | (2u << 30); // addr[56:32] | type=2
    v8i g1;
    g1[0] = (int)((1u << 16) | (1u << 20) | (4u << 22) | (3u << 25)); // f16,pad en,intv,amt
    g1[1] = (int)((rem_k & 0xFFFFu) << 16);                 // tensor_dim0 lo (abar=0)
    g1[2] = (int)((rem_k >> 16) | ((rem_rows & 0xFFFFu) << 16)); // dim0 hi | dim1 lo
    g1[3] = (int)((rem_rows >> 16) | (64u << 16));          // dim1 hi | tile_dim0=64
    g1[4] = 128;                                            // tile_dim1=128, tile_dim2=0
    g1[5] = (int)H_;                                        // tensor_dim0_stride lo
    g1[6] = 0;
    g1[7] = 0;
    v4i g2; g2[0] = g2[1] = g2[2] = g2[3] = 0;
    v4i g3; g3[0] = g3[1] = g3[2] = g3[3] = 0;
#if TDM_6ARG
    v8i g4; g4[0]=g4[1]=g4[2]=g4[3]=g4[4]=g4[5]=g4[6]=g4[7]=0;
    __builtin_amdgcn_tensor_load_to_lds(g0, g1, g2, g3, g4, 0);
#else
    __builtin_amdgcn_tensor_load_to_lds(g0, g1, g2, g3, 0);
#endif
}
#endif // HAVE_TDM

// ============================================================================
// Pass 0: streaming f32 -> f16 convert (8 elems / thread, b128 in, b128 out)
// ============================================================================
__global__ __launch_bounds__(256)
void cvt_f16(const float* __restrict__ src, _Float16* __restrict__ dst, int n) {
    const int i = (blockIdx.x * 256 + threadIdx.x) * 8;
    if (i + 8 <= n) {
        const float4 a = *reinterpret_cast<const float4*>(src + i);
        const float4 b = *reinterpret_cast<const float4*>(src + i + 4);
        v8h h;
        h[0] = (_Float16)a.x; h[1] = (_Float16)a.y;
        h[2] = (_Float16)a.z; h[3] = (_Float16)a.w;
        h[4] = (_Float16)b.x; h[5] = (_Float16)b.y;
        h[6] = (_Float16)b.z; h[7] = (_Float16)b.w;
        *reinterpret_cast<v8h*>(dst + i) = h;
    }
}

// ============================================================================
// Pass 1: logits[128 x 32000] = hidden @ W^T + bias  (f16 WMMA, f32 accum)
// Block: 256 thr / 8 waves. A (f16) fully LDS-resident via async copies;
// W tiles streamed by TDM, double buffered. Wave tile: 32(M) x 64(N).
// ============================================================================
__global__ __launch_bounds__(256)
void hs_gemm(const _Float16* __restrict__ Af16,   // (128,1024) f16
             const _Float16* __restrict__ Wf16,   // (31999,1024) f16
             const float* __restrict__ bias,
             float* __restrict__ logits)
{
    extern __shared__ char smem[];

    const int t    = threadIdx.x;
    const int lane = t & 31;            // wave32
    const int wv   = t >> 5;            // 0..7
    const int wm   = wv & 3;            // M group (32 rows)
    const int wn   = wv >> 2;           // N group (64 cols)
    const int half = lane >> 4;
    const int mr   = lane & 15;
    const int nbase = blockIdx.x * TN;

    const _Float16* wtile0 = Wf16 + (size_t)nbase * H_;
    const unsigned rem_rows = (unsigned)(NW - nbase);   // TDM zero-fills row 31999

    // ---- prologue: kick TDM for stage 0, async-copy whole A into LDS ----
#if HAVE_TDM
    if (wv == 0)
        tdm_load_w_tile(wtile0,
                        (unsigned)(unsigned long long)(smem + W0_OFF),
                        (unsigned)H_, rem_rows);
#else
    { // fallback: per-lane async, 4 x b128 chunks per thread
        #pragma unroll
        for (int j = 0; j < 4; ++j) {
            const int c = t + 256 * j;            // 0..1023
            int row = c >> 3;                     // 0..127
            const int cc = c & 7;                 // 16B chunk in row
            int gr = nbase + row; if (gr > NW - 1) gr = NW - 1;
            async_b128((const char*)Wf16 + (size_t)gr * H_ * 2 + cc * 16,
                       smem + W0_OFF + row * W_STRIDE_B + cc * 16);
        }
    }
#endif
    // A: 128 rows x 128 chunks(16B) = 16384 chunks, 64 per thread
    for (int j = 0; j < 64; ++j) {
        const int c = t + 256 * j;
        const int row = c >> 7;                   // 0..127
        const int cc  = c & 127;                  // 16B chunk
        async_b128((const char*)Af16 + (size_t)row * H_ * 2 + cc * 16,
                   smem + row * A_STRIDE_B + cc * 16);
    }
    wait_async0();
    __syncthreads();

    v8f acc[8];
    #pragma unroll
    for (int i = 0; i < 8; ++i)
        #pragma unroll
        for (int j = 0; j < 8; ++j) acc[i][j] = 0.0f;

    const char* const ab = smem;

    for (int s = 0; s < NSTAGE; ++s) {
        // prefetch stage s+1, then ensure stage s landed
#if HAVE_TDM
        if (wv == 0) {
            if (s + 1 < NSTAGE) {
                tdm_load_w_tile((const char*)wtile0 + (size_t)(s + 1) * KS * 2,
                                (unsigned)(unsigned long long)
                                    (smem + W0_OFF + ((s + 1) & 1) * W_TILE_B),
                                (unsigned)(H_ - (s + 1) * KS), rem_rows);
                __builtin_amdgcn_s_wait_tensorcnt((short)1);  // stage s landed
            } else {
                __builtin_amdgcn_s_wait_tensorcnt((short)0);  // last stage landed
            }
        }
#else
        if (s + 1 < NSTAGE) {
            #pragma unroll
            for (int j = 0; j < 4; ++j) {
                const int c = t + 256 * j;
                int row = c >> 3;
                const int cc = c & 7;
                int gr = nbase + row; if (gr > NW - 1) gr = NW - 1;
                async_b128((const char*)Wf16 + (size_t)gr * H_ * 2
                               + (size_t)(s + 1) * KS * 2 + cc * 16,
                           smem + W0_OFF + ((s + 1) & 1) * W_TILE_B
                               + row * W_STRIDE_B + cc * 16);
            }
            wait_async4();
        } else {
            wait_async0();
        }
#endif
        __syncthreads();

        const char* wb = smem + W0_OFF + (s & 1) * W_TILE_B;
        #pragma unroll
        for (int kk2 = 0; kk2 < 2; ++kk2) {
            const int kbyte = s * (KS * 2) + kk2 * 64;   // k offset in bytes (f16)
            // A fragments (ISA 16-bit A 16x32 layout): lane m=mr,
            // k in [8*half,8*half+8) and [16+8*half,16+8*half+8)
            v16h afr[2];
            #pragma unroll
            for (int mt = 0; mt < 2; ++mt) {
                const int row = wm * 32 + mt * 16 + mr;
                const char* p = ab + (size_t)row * A_STRIDE_B + kbyte + half * 16;
                const v8h lo = *reinterpret_cast<const v8h*>(p);
                const v8h hi = *reinterpret_cast<const v8h*>(p + 32);
                afr[mt] = __builtin_shufflevector(lo, hi,
                    0,1,2,3,4,5,6,7,8,9,10,11,12,13,14,15);
            }
            #pragma unroll
            for (int nt = 0; nt < 4; ++nt) {
                // B fragment: lane n=mr, k in [8*half,8*half+8) and [16+8*half,16+8*half+8)
                const int row = wn * 64 + nt * 16 + mr;
                const char* p = wb + (size_t)row * W_STRIDE_B + kk2 * 64 + half * 16;
                const v8h lo = *reinterpret_cast<const v8h*>(p);
                const v8h hi = *reinterpret_cast<const v8h*>(p + 32);
                const v16h bfr = __builtin_shufflevector(lo, hi,
                    0,1,2,3,4,5,6,7,8,9,10,11,12,13,14,15);
                acc[0 * 4 + nt] = __builtin_amdgcn_wmma_f32_16x16x32_f16(
                    false, afr[0], false, bfr, (short)0, acc[0 * 4 + nt], false, false);
                acc[1 * 4 + nt] = __builtin_amdgcn_wmma_f32_16x16x32_f16(
                    false, afr[1], false, bfr, (short)0, acc[1 * 4 + nt], false, false);
            }
        }
        __syncthreads();   // current W buffer free for TDM issued next iteration
    }

    // epilogue: C/D layout (M = 8*half + r, N = mr) per accumulator VGPR r
    #pragma unroll
    for (int mt = 0; mt < 2; ++mt) {
        #pragma unroll
        for (int nt = 0; nt < 4; ++nt) {
            const int n = nbase + wn * 64 + nt * 16 + mr;
            const float bb = (n < NW) ? bias[n] : 0.0f;
            #pragma unroll
            for (int r = 0; r < 8; ++r) {
                const int m = wm * 32 + mt * 16 + half * 8 + r;
                logits[(size_t)m * V_ + n] = acc[mt * 4 + nt][r] + bb;
            }
        }
    }
}

// ============================================================================
// Pass 2: out[b, tok] = sum_d logsigmoid(sign_d * logits[b, node_d])
// Thread = (token, batch-chunk of 16); logits (16 MB) are L2-resident.
// ============================================================================
__global__ __launch_bounds__(256)
void hs_path(const float* __restrict__ logits,
             const int* __restrict__ nodes,
             const int* __restrict__ signs,
             float* __restrict__ out, int D)
{
    const int i   = blockIdx.x * 256 + threadIdx.x;   // over (B/16)*V
    const int bc  = i / V_;
    const int tok = i - bc * V_;

    int   nd[MAXD];
    float sg[MAXD];
    int De = 0;
    for (int d = 0; d < D; ++d) {
        const int s = signs[tok * D + d];
        if (s == 0) break;                  // zero-sign padding trails
        nd[De] = nodes[tok * D + d];
        sg[De] = (float)s;
        ++De;
    }

    #pragma unroll 4
    for (int b2 = 0; b2 < 16; ++b2) {
        const int b = bc * 16 + b2;
        const float* Lr = logits + (size_t)b * V_;
        float acc = 0.0f;
        for (int d = 0; d < De; ++d) {
            const float x  = Lr[nd[d]] * sg[d];
            const float mn = fminf(x, 0.0f);
            acc += mn - __logf(1.0f + __expf(-fabsf(x)));
        }
        out[(size_t)b * V_ + tok] = acc;
    }
}

extern "C" void kernel_launch(void* const* d_in, const int* in_sizes, int n_in,
                              void* d_out, int out_size, void* d_ws, size_t ws_size,
                              hipStream_t stream) {
    const float* hidden = (const float*)d_in[0];   // (128, 1024) f32
    const float* W      = (const float*)d_in[1];   // (31999, 1024) f32
    const float* bias   = (const float*)d_in[2];   // (31999,) f32
    const int*   nodes  = (const int*)d_in[3];     // (32000, D) i32
    const int*   signs  = (const int*)d_in[4];     // (32000, D) i32
    float*       out    = (float*)d_out;           // (128, 32000) f32

    // workspace: [logits f32 16.384 MB][W f16 65.534 MB][A f16 256 KB]
    char* ws = (char*)d_ws;
    float*     logits = (float*)ws;
    _Float16*  Wf16   = (_Float16*)(ws + (size_t)B_ * V_ * 4);
    _Float16*  Af16   = (_Float16*)(ws + (size_t)B_ * V_ * 4 + (size_t)NW * H_ * 2);

    int D = in_sizes[3] / V_;
    if (D > MAXD) D = MAXD;

    const int nW = NW * H_;                 // 32,766,976 (divisible by 8)
    const int nA = B_ * H_;                 // 131,072
    cvt_f16<<<(nW / 8 + 255) / 256, 256, 0, stream>>>(W, Wf16, nW);
    cvt_f16<<<(nA / 8 + 255) / 256, 256, 0, stream>>>(hidden, Af16, nA);
    hs_gemm<<<(NW + TN - 1) / TN, 256, SMEM_BYTES, stream>>>(Af16, Wf16, bias, logits);
    hs_path<<<((B_ / 16) * V_) / 256, 256, 0, stream>>>(logits, nodes, signs, out, D);
}
